// SelectedUnitsHead_42838003810505
// MI455X (gfx1250) — matrix operand order repaired and checked by reading, not compiled, not run on hardware
//
#include <hip/hip_runtime.h>
#include <hip/hip_bf16.h>

// ---------------------------------------------------------------------------
// SelectedUnitsHead for MI455X (gfx1250), wave32 + WMMA bf16 (f32 accumulate).
//
// Kernel 0: pack weights f32 -> bf16 in WMMA B-operand register layout.
// Kernel 1: key_proj = entity_embeddings @ Wk + bk  (HBM-bound, whole chip).
// Kernel 2: fused 64-step autoregressive loop. 16 batch rows / WG (WMMA M=16),
//           16 WGs x 512 thr. All state in LDS (~136KB of 320KB/WGP); bf16
//           A-layout shadows so every WMMA operand is one 32B load per lane.
// ---------------------------------------------------------------------------

typedef __attribute__((ext_vector_type(16))) __bf16 v16bf;
typedef __attribute__((ext_vector_type(8)))  float  v8f;

static constexpr int Bc   = 256;   // batch
static constexpr int Ec   = 512;   // entities
static constexpr int Cc   = 256;   // channels
static constexpr int ARc  = 1024;  // autoregressive dim
static constexpr int H256 = 256;
static constexpr int H32  = 32;
static constexpr int NSEL = 64;

// packed-weight offsets in d_ws (in halves), after the f32 key_proj buffer
static constexpr long long KP_F32   = (long long)Bc * Ec * H32;  // floats
static constexpr long long WFP_OFF  = 0;
static constexpr long long W1P_OFF  = WFP_OFF + 256 * 256;
static constexpr long long W2P_OFF  = W1P_OFF + 1024 * 256;
static constexpr long long WXP_OFF  = W2P_OFF + 256 * 32;
static constexpr long long WHP_OFF  = WXP_OFF + 32 * 128;
static constexpr long long WPP_OFF  = WHP_OFF + 32 * 128;
static constexpr long long WKP_OFF  = WPP_OFF + 32 * 1024;

__device__ __forceinline__ v8f wmma_bf16(v16bf a, v16bf b, v8f c) {
  return __builtin_amdgcn_wmma_f32_16x16x32_bf16(false, a, false, b, (short)0, c,
                                                 false, false);
}

// ---- A-matrix 16x32 bf16 register layout (ISA 7.12.2) -----------------------
// lane = g*16+m ; halves e=2v+j ; K = 2v+j + 8g + (v>=4 ? 8 : 0)
// inverse: given kk=K&31 : g=(kk>>3)&1 ; e=(kk&7)+((kk&16)?8:0)

// gather-load A from f32 (used once for the one-hot table GEMM)
__device__ __forceinline__ v16bf load_a16x32_rows(const float* table, int ld,
                                                  const int* rows, int k0) {
  int lane = threadIdx.x & 31;
  int m = lane & 15, g = lane >> 4;
  const float* src = table + (long long)rows[m] * ld;
  v16bf a;
#pragma unroll
  for (int v = 0; v < 8; ++v)
#pragma unroll
    for (int j = 0; j < 2; ++j) {
      int kk = 2 * v + j + 8 * g + ((v >= 4) ? 8 : 0);
      a[2 * v + j] = (__bf16)src[k0 + kk];
    }
  return a;
}

// contiguous-f32 A load (key_proj: rows of entity_embeddings)
__device__ __forceinline__ v16bf load_a16x32(const float* src, int ld, int k0) {
  int lane = threadIdx.x & 31;
  int m = lane & 15, g = lane >> 4;
  v16bf a;
#pragma unroll
  for (int v = 0; v < 8; ++v)
#pragma unroll
    for (int j = 0; j < 2; ++j) {
      int kk = 2 * v + j + 8 * g + ((v >= 4) ? 8 : 0);
      a[2 * v + j] = (__bf16)src[m * ld + k0 + kk];
    }
  return a;
}

// packed A shadow: tile kt -> 32 lanes x 16 halves, one 32B load per lane
__device__ __forceinline__ v16bf load_ap(const __bf16* ap, int kt) {
  int lane = threadIdx.x & 31;
  return *(const v16bf*)(ap + ((kt * 32 + lane) << 4));
}
// store one element (m,k) into packed A shadow
__device__ __forceinline__ void pack_a_store(__bf16* ap, int m, int k, float v) {
  int kt = k >> 5, kk = k & 31;
  int g = (kk >> 3) & 1;
  int e = (kk & 7) + ((kk & 16) ? 8 : 0);
  ap[((kt * 32 + g * 16 + m) << 4) + e] = (__bf16)v;
}

// packed B weight: tile (kt,nt) -> 32 lanes x 16 halves, one 32B load per lane
__device__ __forceinline__ v16bf load_bp(const __bf16* wp, int NT, int kt, int nt) {
  int lane = threadIdx.x & 31;
  return *(const v16bf*)(wp + (((kt * NT + nt) * 32 + lane) << 4));
}

__device__ __forceinline__ float sigmoidf_(float x) {
  return 1.0f / (1.0f + __expf(-x));
}

// Counter-based Gumbel noise (jax PRNG stream not reproducible; deterministic).
__device__ __forceinline__ float gumbelf(int t, int b, int e) {
  unsigned x = (unsigned)t * 0x9E3779B1u + (unsigned)b * 0x85EBCA77u +
               (unsigned)e * 0xC2B2AE3Du + 0x27220A95u;
  x ^= x >> 16; x *= 0x7FEB352Du;
  x ^= x >> 15; x *= 0x846CA68Bu;
  x ^= x >> 16;
  float u = fmaxf((float)(x >> 8) * (1.0f / 16777216.0f), 1e-20f);
  return -__logf(-__logf(u));
}

// ---------------------------------------------------------------------------
// Kernel 0: pack a K x N f32 weight into bf16 WMMA B-tile layout.
// dst half index = ((kt*NT + nt)*32 + lane)*16 + e ; K = 2v+j+8g+(v>=4?8:0), N = nt*16+n
// ---------------------------------------------------------------------------
__global__ void pack_b_kernel(const float* __restrict__ src,
                              __bf16* __restrict__ dst, int K, int N) {
  int idx = blockIdx.x * blockDim.x + threadIdx.x;
  if (idx >= K * N) return;
  int e    = idx & 15;
  int lane = (idx >> 4) & 31;
  int tile = idx >> 9;
  int NT = N >> 4;
  int kt = tile / NT, nt = tile - kt * NT;
  int n = lane & 15, g = lane >> 4;
  int v = e >> 1, j = e & 1;
  int kk = 2 * v + j + 8 * g + ((v >= 4) ? 8 : 0);
  dst[idx] = (__bf16)src[(kt * 32 + kk) * N + nt * 16 + n];
}

// ---------------------------------------------------------------------------
// Kernel 1: key_proj[b,e,:] = ent[b,e,:] @ Wk + bk.  131072x(K=256,N=32)
// ---------------------------------------------------------------------------
__global__ void keyproj_kernel(const float* __restrict__ ent,
                               const __bf16* __restrict__ Wkp,
                               const float* __restrict__ bk,
                               float* __restrict__ kp) {
  int wave = blockIdx.x * (blockDim.x >> 5) + (threadIdx.x >> 5);
  long long row0 = (long long)wave * 16;
  const float* a0 = ent + row0 * Cc;
  v8f c0 = {}; v8f c1 = {};
  for (int k0 = 0; k0 < Cc; k0 += 32) {
    __builtin_prefetch(a0 + (k0 + 64), 0, 1);  // global_prefetch_b8
    v16bf a  = load_a16x32(a0, Cc, k0);
    v16bf b0 = load_bp(Wkp, 2, k0 >> 5, 0);
    v16bf b1 = load_bp(Wkp, 2, k0 >> 5, 1);
    c0 = wmma_bf16(a, b0, c0);
    c1 = wmma_bf16(a, b1, c1);
  }
  int lane = threadIdx.x & 31;
  int n = lane & 15, g = lane >> 4;
#pragma unroll
  for (int r = 0; r < 8; ++r) {
    long long m = row0 + r + 8 * g;
    kp[m * H32 + n]      = c0[r] + bk[n];
    kp[m * H32 + 16 + n] = c1[r] + bk[16 + n];
  }
}

// ---------------------------------------------------------------------------
// Kernel 2: fused autoregressive selection loop. 16 batch rows per workgroup.
// ---------------------------------------------------------------------------
__global__ void __launch_bounds__(512)
selunits_kernel(const float* __restrict__ ar_in,
                const long long* __restrict__ act,
                const float* __restrict__ umask,
                const float* __restrict__ stab,
                const float* __restrict__ bfv, const float* __restrict__ b1,
                const float* __restrict__ b2,  const float* __restrict__ bl,
                const float* __restrict__ bp,
                const __bf16* __restrict__ Wfp, const __bf16* __restrict__ W1p,
                const __bf16* __restrict__ W2p, const __bf16* __restrict__ Wxp,
                const __bf16* __restrict__ Whp, const __bf16* __restrict__ Wpp,
                const float* __restrict__ kp,
                float* __restrict__ out) {
  extern __shared__ unsigned char smem_raw[];
  float* ar_s  = (float*)smem_raw;            // 16 x 1024 f32 master
  float* fe_s  = ar_s + 16 * ARc;             // 16 x 256
  float* zg_s  = fe_s + 16 * H256;            // 16 x 128
  float* h_s   = zg_s + 16 * 128;             // 16 x 32 (f32, for attention)
  float* c_s   = h_s + 16 * H32;              // 16 x 32
  float* sel_s = c_s + 16 * H32;              // 16
  __bf16* arp   = (__bf16*)(sel_s + 16);      // packed A: 16 x 1024
  __bf16* xactp = arp + 16 * ARc;             // packed A: 16 x 256
  __bf16* z2p   = xactp + 16 * H256;          // packed A: 16 x 32
  __bf16* hp    = z2p + 16 * H32;             // packed A: 16 x 32
  __bf16* outp  = hp + 16 * H32;              // packed A: 16 x 32
  unsigned* maskw = (unsigned*)(outp + 16 * H32);  // 16 x 16 bit-words
  int* actions = (int*)(maskw + 16 * 16);     // 16
  int* eid_s   = actions + 16;                // 16

  const int tid  = threadIdx.x;
  const int w    = tid >> 5;
  const int lane = tid & 31;
  const int nn   = lane & 15, gg = lane >> 4;
  const int b0   = blockIdx.x * 16;

  const long long UOFF = (long long)Bc * NSEL * Ec;
  const long long AOFF = UOFF + (long long)Bc * NSEL;

  // -------- prologue --------
  if (tid < 16) {
    int a = (int)act[b0 + tid];
    actions[tid] = a;
    sel_s[tid]   = stab[a];
  }
  __syncthreads();

  // the_func_embed = relu(gather(umask, a) @ Wf + bf): wave w -> cols w*16..
  {
    int n0 = w * 16;
    v8f acc = {};
    for (int k0 = 0; k0 < H256; k0 += 32) {
      v16bf a = load_a16x32_rows(umask, H256, actions, k0);
      v16bf b = load_bp(Wfp, 16, k0 >> 5, w);
      acc = wmma_bf16(a, b, acc);
    }
#pragma unroll
    for (int r = 0; r < 8; ++r) {
      int mm = r + 8 * gg;
      fe_s[mm * H256 + n0 + nn] = fmaxf(acc[r] + bfv[n0 + nn], 0.0f);
    }
  }
  // ar (f32 master + packed bf16 shadow), h, c, mask init
  for (int idx = tid; idx < 16 * ARc; idx += 512) {
    float v = ar_in[(long long)b0 * ARc + idx];
    ar_s[idx] = v;
    pack_a_store(arp, idx >> 10, idx & (ARc - 1), v);
  }
  { h_s[tid] = 0.0f; c_s[tid] = 0.0f; hp[tid] = (__bf16)0.0f; }
  if (tid < 256) maskw[tid] = 0xFFFFFFFFu;
  __syncthreads();

  // -------- 64 sequential selection steps --------
  for (int t = 0; t < NSEL; ++t) {
    // Phase 1: x = ar @ W1 + b1 ; xact = relu(fe + x) -> packed shadow only.
    {
      int n0 = w * 16;
      v8f acc = {};
      for (int kt = 0; kt < ARc / 32; ++kt) {
        v16bf a = load_ap(arp, kt);           // one 32B LDS load / lane
        v16bf b = load_bp(W1p, 16, kt, w);    // one 32B global load / lane
        acc = wmma_bf16(a, b, acc);
      }
#pragma unroll
      for (int r = 0; r < 8; ++r) {
        int mm = r + 8 * gg;
        float xv = acc[r] + b1[n0 + nn] + fe_s[mm * H256 + n0 + nn];
        pack_a_store(xactp, mm, n0 + nn, fmaxf(xv, 0.0f));
      }
    }
    __syncthreads();

    // Phase 2: waves 0-1: z2 = xact @ W2 + b2. waves 2-9: zg = h @ Wh + bl.
    if (w < 2) {
      v8f acc = {};
      for (int kt = 0; kt < H256 / 32; ++kt) {
        v16bf a = load_ap(xactp, kt);
        v16bf b = load_bp(W2p, 2, kt, w);
        acc = wmma_bf16(a, b, acc);
      }
      int n0 = w * 16;
#pragma unroll
      for (int r = 0; r < 8; ++r) {
        int mm = r + 8 * gg;
        pack_a_store(z2p, mm, n0 + nn, acc[r] + b2[n0 + nn]);
      }
    } else if (w < 10) {
      int wv = w - 2, n0 = wv * 16;
      v16bf a = load_ap(hp, 0);
      v16bf b = load_bp(Whp, 8, 0, wv);
      v8f z = {};
      z = wmma_bf16(a, b, z);
#pragma unroll
      for (int r = 0; r < 8; ++r) {
        int mm = r + 8 * gg;
        zg_s[mm * 128 + n0 + nn] = z[r] + bl[n0 + nn];
      }
    }
    __syncthreads();

    // Phase 3: zg += z2 @ Wx (accumulate into existing C).
    if (w < 8) {
      int n0 = w * 16;
      v8f cin;
#pragma unroll
      for (int r = 0; r < 8; ++r) cin[r] = zg_s[(r + 8 * gg) * 128 + n0 + nn];
      v16bf a = load_ap(z2p, 0);
      v16bf b = load_bp(Wxp, 8, 0, w);
      v8f z = wmma_bf16(a, b, cin);
#pragma unroll
      for (int r = 0; r < 8; ++r) zg_s[(r + 8 * gg) * 128 + n0 + nn] = z[r];
    }
    __syncthreads();

    // Phase 4: LSTM gates, 16 rows x 32 units = 512 lanes.
    {
      int mm = tid >> 5, u = tid & 31;
      float zi = zg_s[mm * 128 + u];
      float zf = zg_s[mm * 128 + 32 + u];
      float zc = zg_s[mm * 128 + 64 + u];
      float zo = zg_s[mm * 128 + 96 + u];
      float i = sigmoidf_(zi), f = sigmoidf_(zf), o = sigmoidf_(zo);
      float cn = f * c_s[mm * H32 + u] + i * tanhf(zc);
      float hv = o * tanhf(cn);
      c_s[mm * H32 + u] = cn;
      h_s[mm * H32 + u] = hv;
      pack_a_store(hp, mm, u, hv);
    }
    __syncthreads();

    // Phase 5: attention logits + softmax + gumbel argmax. wave w -> row w.
    {
      int mm = w;
      long long brow = (long long)(b0 + mm);
      const float* kprow = kp + brow * (Ec * H32);
      float sel = sel_s[mm];
      float yv[16], rowmax = -3.4e38f;
#pragma unroll
      for (int ii = 0; ii < 16; ++ii) {
        int e = lane + ii * 32;
        const float4* kp4 = (const float4*)(kprow + (long long)e * H32);
        float acc = 0.0f;
#pragma unroll
        for (int q = 0; q < 8; ++q) {
          float4 kv = kp4[q];
          acc += kv.x * h_s[mm * H32 + 4 * q + 0];
          acc += kv.y * h_s[mm * H32 + 4 * q + 1];
          acc += kv.z * h_s[mm * H32 + 4 * q + 2];
          acc += kv.w * h_s[mm * H32 + 4 * q + 3];
        }
        float mk = (float)((maskw[mm * 16 + (e >> 5)] >> (e & 31)) & 1u);
        float y2 = acc * mk;        // TEMPERATURE == 1
        yv[ii] = y2;
        rowmax = fmaxf(rowmax, y2);
        out[(brow * NSEL + t) * Ec + e] = y2 * sel;
      }
#pragma unroll
      for (int off = 16; off; off >>= 1)
        rowmax = fmaxf(rowmax, __shfl_xor(rowmax, off, 32));
      float pexp[16], ssum = 0.0f;
#pragma unroll
      for (int ii = 0; ii < 16; ++ii) { pexp[ii] = __expf(yv[ii] - rowmax); ssum += pexp[ii]; }
#pragma unroll
      for (int off = 16; off; off >>= 1) ssum += __shfl_xor(ssum, off, 32);
      float inv = 1.0f / ssum;
      float best = -3.4e38f; int bi = 0;
#pragma unroll
      for (int ii = 0; ii < 16; ++ii) {
        int e = lane + ii * 32;
        float s = pexp[ii] * inv + gumbelf(t, (int)brow, e);
        if (s > best) { best = s; bi = e; }
      }
#pragma unroll
      for (int off = 16; off; off >>= 1) {
        float ob = __shfl_xor(best, off, 32);
        int   oi = __shfl_xor(bi, off, 32);
        if (ob > best || (ob == best && oi < bi)) { best = ob; bi = oi; }
      }
      if (lane == 0) {
        eid_s[mm] = bi;
        maskw[mm * 16 + (bi >> 5)] &= ~(1u << (bi & 31));
        out[UOFF + brow * NSEL + t] = (float)bi * sel;
      }
      // out = key_proj[b, eid, :] - mean(one_hot) (= 1/E); packed A shadow
      float ov = kprow[(long long)bi * H32 + lane] - (1.0f / (float)Ec);
      pack_a_store(outp, mm, lane, ov);
    }
    __syncthreads();

    // Phase 6: ar += out @ Wp + bp. wave w -> 4 N-tiles of 16 (K=32, 1 wmma).
    {
      v16bf a = load_ap(outp, 0);
#pragma unroll
      for (int s4 = 0; s4 < 4; ++s4) {
        int nt = w * 4 + s4, n0 = nt * 16;
        v16bf b = load_bp(Wpp, 64, 0, nt);
        v8f z = {};
        z = wmma_bf16(a, b, z);
#pragma unroll
        for (int r = 0; r < 8; ++r) {
          int mm = r + 8 * gg;
          float nv = ar_s[mm * ARc + n0 + nn] + z[r] + bp[n0 + nn];
          ar_s[mm * ARc + n0 + nn] = nv;
          pack_a_store(arp, mm, n0 + nn, nv);
        }
      }
    }
    __syncthreads();
  }

  // -------- epilogue: ar * select --------
  for (int idx = tid; idx < 16 * ARc; idx += 512) {
    int mm = idx >> 10;
    out[AOFF + (long long)b0 * ARc + idx] = ar_s[idx] * sel_s[mm];
  }
}

// ---------------------------------------------------------------------------
extern "C" void kernel_launch(void* const* d_in, const int* in_sizes, int n_in,
                              void* d_out, int out_size, void* d_ws, size_t ws_size,
                              hipStream_t stream) {
  const float*     ar_in = (const float*)d_in[0];
  const long long* act   = (const long long*)d_in[1];   // int64 action ids
  const float*     ent   = (const float*)d_in[2];
  const float*     Wf    = (const float*)d_in[3];
  const float*     bfv   = (const float*)d_in[4];
  const float*     Wk    = (const float*)d_in[5];
  const float*     bk    = (const float*)d_in[6];
  const float*     W1    = (const float*)d_in[7];
  const float*     b1    = (const float*)d_in[8];
  const float*     W2    = (const float*)d_in[9];
  const float*     b2    = (const float*)d_in[10];
  const float*     Wx    = (const float*)d_in[11];
  const float*     Wh    = (const float*)d_in[12];
  const float*     bl    = (const float*)d_in[13];
  const float*     Wp    = (const float*)d_in[14];
  const float*     bp    = (const float*)d_in[15];
  const float*     umask = (const float*)d_in[16];
  const float*     stab  = (const float*)d_in[17];
  float* out = (float*)d_out;

  float*  kp = (float*)d_ws;                 // B*E*H32 f32 = 16 MB
  __bf16* wp = (__bf16*)(kp + KP_F32);       // packed bf16 weights (~770 KB)

  // Kernel 0: pack weights (idempotent, graph-capture safe)
  auto pack = [&](const float* src, long long off, int K, int N) {
    int n = K * N;
    pack_b_kernel<<<(n + 255) / 256, 256, 0, stream>>>(src, wp + off, K, N);
  };
  pack(Wf, WFP_OFF, 256, 256);
  pack(W1, W1P_OFF, 1024, 256);
  pack(W2, W2P_OFF, 256, 32);
  pack(Wx, WXP_OFF, 32, 128);
  pack(Wh, WHP_OFF, 32, 128);
  pack(Wp, WPP_OFF, 32, 1024);
  pack(Wk, WKP_OFF, 256, 32);

  // Kernel 1: key_proj — 131072 rows / 16 per wave / 8 waves per block
  keyproj_kernel<<<1024, 256, 0, stream>>>(ent, wp + WKP_OFF, bk, kp);

  // Kernel 2: fused loop
  size_t smem_f32 = (size_t)(16 * ARc + 16 * H256 + 16 * 128 + 16 * H32 +
                             16 * H32 + 16) * 4;
  size_t smem_bf  = (size_t)(16 * ARc + 16 * H256 + 3 * 16 * H32) * 2;
  size_t smem_int = (size_t)(16 * 16 + 16 + 16) * 4;
  size_t smem = smem_f32 + smem_bf + smem_int;
  (void)hipFuncSetAttribute((const void*)selunits_kernel,
                            hipFuncAttributeMaxDynamicSharedMemorySize,
                            (int)smem);
  selunits_kernel<<<16, 512, smem, stream>>>(
      ar_in, act, umask, stab, bfv, b1, b2, bl, bp,
      wp + WFP_OFF, wp + W1P_OFF, wp + W2P_OFF, wp + WXP_OFF, wp + WHP_OFF,
      wp + WPP_OFF, kp, out);
}
